// MixtureOfExpertsLayer_53558242181864
// MI455X (gfx1250) — matrix operand order repaired and checked by reading, not compiled, not run on hardware
//
#include <hip/hip_runtime.h>
#include <hip/hip_bf16.h>
#include <math.h>

// ---------- types ----------
typedef __bf16 bf16_t;
typedef __attribute__((ext_vector_type(16))) __bf16 v16bf;
typedef __attribute__((ext_vector_type(8)))  __bf16 v8bf;
typedef __attribute__((ext_vector_type(4)))  __bf16 v4bf;
typedef __attribute__((ext_vector_type(8)))  float  v8f;
typedef __attribute__((ext_vector_type(4)))  int    v4i;

#if __has_builtin(__builtin_amdgcn_global_load_async_to_lds_b128)
#define USE_ASYNC_LDS 1
typedef __attribute__((address_space(1))) v4i* gas_v4i;
typedef __attribute__((address_space(3))) v4i* las_v4i;
#else
#define USE_ASYNC_LDS 0
#endif

// ---------- problem constants (fixed by reference) ----------
constexpr int Bn = 4;
constexpr int Ln = 4096;
constexpr int Dn = 1024;
constexpr int En = 8;
constexpr int Tn = Bn * Ln;            // 16384 tokens
constexpr int NSLOTS = 2 * Tn;         // 32768 (token,expert) assignments
constexpr int PAD_SLOTS = 33792;       // 32768 + 8*128 padding, 128-aligned
constexpr int MAXB = PAD_SLOTS / 128;  // 264 M-blocks worst case

// GEMM tiling: block tile 128(M) x 128(N), K chunk 64, double-buffered LDS
constexpr int KC  = 64;
constexpr int KCP = KC + 8;            // 72 bf16 = 144B row stride (16B aligned, bank-spread)
constexpr int NC  = Dn / KC;           // 16 K-chunks

// ---------- workspace layout (bytes) ----------
constexpr size_t OFF_TOPW    = 0;                          // [2T] f32
constexpr size_t OFF_TOPI    = OFF_TOPW + 2ull*Tn*4;       // [2T] i32
constexpr size_t OFF_T2S     = OFF_TOPI + 2ull*Tn*4;       // [2T] i32
constexpr size_t OFF_SW      = OFF_T2S  + 2ull*Tn*4;       // [PAD_SLOTS] f32
constexpr size_t OFF_COUNTS  = OFF_SW   + (size_t)PAD_SLOTS*4; // 8 i32
constexpr size_t OFF_CURSORS = OFF_COUNTS + 256;           // 8 i32
constexpr size_t OFF_BASES   = OFF_CURSORS + 256;          // 8 i32
constexpr size_t OFF_BEXP    = OFF_BASES + 256;            // MAXB i32
constexpr size_t OFF_BSLOT   = OFF_BEXP  + 2048;
constexpr size_t OFF_BROWS   = OFF_BSLOT + 2048;
constexpr size_t OFF_XG      = 589824;                               // [PAD_SLOTS*D] bf16
constexpr size_t OFF_HG      = OFF_XG + (size_t)PAD_SLOTS*Dn*2;      // [PAD_SLOTS*D] bf16
constexpr size_t OFF_YG      = OFF_HG + (size_t)PAD_SLOTS*Dn*2;      // [PAD_SLOTS*D] f32
constexpr size_t OFF_W1B     = OFF_YG + (size_t)PAD_SLOTS*Dn*4;      // [E*D*D] bf16
constexpr size_t OFF_W2B     = OFF_W1B + (size_t)En*Dn*Dn*2;         // [E*D*D] bf16

// ---------- kernel 0: zero counters ----------
__global__ void moe_init(int* __restrict__ counts, int* __restrict__ cursors) {
    if (threadIdx.x < En) { counts[threadIdx.x] = 0; cursors[threadIdx.x] = 0; }
}

// ---------- kernel 0b: one-time weight convert fp32 -> bf16 ----------
__global__ __launch_bounds__(256) void moe_cvtw(const float* __restrict__ src,
                                                bf16_t* __restrict__ dst) {
    size_t i = (size_t)(blockIdx.x * 256 + threadIdx.x) * 8;
    float4 a = *(const float4*)(src + i);
    float4 b = *(const float4*)(src + i + 4);
    v8bf o = { (bf16_t)a.x, (bf16_t)a.y, (bf16_t)a.z, (bf16_t)a.w,
               (bf16_t)b.x, (bf16_t)b.y, (bf16_t)b.z, (bf16_t)b.w };
    *(v8bf*)(dst + i) = o;
}

// ---------- kernel 1: router (one wave per token) ----------
__global__ __launch_bounds__(256) void moe_router(
    const float* __restrict__ x, const float* __restrict__ Wr,
    float* __restrict__ topw, int* __restrict__ topi, int* __restrict__ counts)
{
    const int lane = threadIdx.x & 31;
    const int wave = threadIdx.x >> 5;
    const int t = blockIdx.x * 8 + wave;
    if (t >= Tn) return;
    const float* xr = x + (size_t)t * Dn;

    float acc[En];
#pragma unroll
    for (int e = 0; e < En; ++e) acc[e] = 0.f;

    for (int j = 0; j < Dn / 32; ++j) {
        float xv = xr[lane + 32 * j];
#pragma unroll
        for (int e = 0; e < En; ++e)
            acc[e] += xv * Wr[(size_t)e * Dn + lane + 32 * j];
    }
    // wave32 cross-lane reduction
#pragma unroll
    for (int e = 0; e < En; ++e)
        for (int off = 16; off > 0; off >>= 1)
            acc[e] += __shfl_xor(acc[e], off, 32);

    if (lane == 0) {
        int i0 = 0; float v0 = acc[0];
#pragma unroll
        for (int e = 1; e < En; ++e) if (acc[e] > v0) { v0 = acc[e]; i0 = e; }
        int i1 = -1; float v1 = -3.0e38f;
#pragma unroll
        for (int e = 0; e < En; ++e) if (e != i0 && acc[e] > v1) { v1 = acc[e]; i1 = e; }
        // renormalized top-2 softmax weights: denominator cancels
        float w0 = 1.f / (1.f + __expf(v1 - v0));
        float w1 = 1.f - w0;
        topi[2 * t + 0] = i0; topi[2 * t + 1] = i1;
        topw[2 * t + 0] = w0; topw[2 * t + 1] = w1;
        atomicAdd(&counts[i0], 1);
        atomicAdd(&counts[i1], 1);
    }
}

// ---------- kernel 2: prefix scan + block tables (1 thread) ----------
__global__ void moe_scan(const int* __restrict__ counts, int* __restrict__ bases,
                         int* __restrict__ blkExp, int* __restrict__ blkSlot,
                         int* __restrict__ blkRows)
{
    if (threadIdx.x != 0 || blockIdx.x != 0) return;
    int base = 0, b = 0;
    for (int e = 0; e < En; ++e) {
        bases[e] = base;
        int c = counts[e];
        int nb = (c + 127) >> 7;
        for (int i = 0; i < nb; ++i) {
            blkExp[b] = e;
            blkSlot[b] = base + i * 128;
            int rem = c - i * 128;
            blkRows[b] = rem < 128 ? rem : 128;
            ++b;
        }
        base += nb << 7;
    }
    for (; b < MAXB; ++b) blkExp[b] = -1;
}

// ---------- kernel 3: gather + fp32->bf16 (one wave per (token,k)) ----------
__global__ __launch_bounds__(256) void moe_gather(
    const float* __restrict__ x, const int* __restrict__ topi,
    const float* __restrict__ topw, const int* __restrict__ bases,
    int* __restrict__ cursors, int* __restrict__ t2s,
    float* __restrict__ slotW, bf16_t* __restrict__ xg)
{
    const int lane = threadIdx.x & 31;
    const int wave = threadIdx.x >> 5;
    const int item = blockIdx.x * 8 + wave;
    if (item >= NSLOTS) return;
    const int t = item >> 1;
    const int e = topi[item];

    int slot = 0;
    if (lane == 0) slot = bases[e] + atomicAdd(&cursors[e], 1);
    slot = __shfl(slot, 0, 32);
    if (lane == 0) { t2s[item] = slot; slotW[slot] = topw[item]; }

    const float4* src = (const float4*)(x + (size_t)t * Dn);
    bf16_t* dst = xg + (size_t)slot * Dn;
#pragma unroll
    for (int i = 0; i < Dn / (32 * 4); ++i) {
        float4 f = src[lane + 32 * i];
        v4bf o = { (bf16_t)f.x, (bf16_t)f.y, (bf16_t)f.z, (bf16_t)f.w };
        *(v4bf*)(dst + 4 * (lane + 32 * i)) = o;
    }
}

// ---------- staging helpers ----------
__device__ __forceinline__ void stage16(const bf16_t* g, bf16_t* l) {
#if USE_ASYNC_LDS
    __builtin_amdgcn_global_load_async_to_lds_b128(
        (gas_v4i)(g), (las_v4i)(l), 0, 0);
#else
    *(v8bf*)l = *(const v8bf*)g;
#endif
}

// wait until at most the 8 most recent async copies remain outstanding
// (async ops complete in issue order -> previous chunk's 8 copies are done)
__device__ __forceinline__ void wait_async_keep8() {
#if USE_ASYNC_LDS
#if __has_builtin(__builtin_amdgcn_s_wait_asynccnt)
    __builtin_amdgcn_s_wait_asynccnt(8);
#else
    asm volatile("s_wait_asynccnt 8" ::: "memory");
#endif
#endif
}
__device__ __forceinline__ void wait_async_all() {
#if USE_ASYNC_LDS
#if __has_builtin(__builtin_amdgcn_s_wait_asynccnt)
    __builtin_amdgcn_s_wait_asynccnt(0);
#else
    asm volatile("s_wait_asynccnt 0" ::: "memory");
#endif
#endif
}

// ---------- kernels 4/5: expert NT GEMM with bf16 WMMA ----------
// D[m,n] = sum_k A[m,k] * W[n,k]  (+bias, +gelu or *slotWeight)
// Block: 256 thr = 8 waves laid out 4(M) x 2(N); wave tile 32M x 64N (2x4 WMMA accums)
// Double-buffered LDS; async copy for chunk c+1 in flight during WMMAs on chunk c.
template<bool GELU, bool SCALE, typename OutT>
__global__ __launch_bounds__(256) void moe_gemm(
    const bf16_t* __restrict__ A,     // [PAD_SLOTS, D] bf16
    const bf16_t* __restrict__ W,     // [E, D, D] bf16 (n-major, k contiguous)
    const float*  __restrict__ bias,  // [E, D]
    const int* __restrict__ blkExp, const int* __restrict__ blkSlot,
    const int* __restrict__ blkRows, const float* __restrict__ slotW,
    OutT* __restrict__ Out)           // [PAD_SLOTS, D]
{
    const int e = blkExp[blockIdx.x];
    if (e < 0) return;
    const int slot0 = blkSlot[blockIdx.x];
    const int rows  = blkRows[blockIdx.x];
    const int nBase = blockIdx.y * 128;

    __shared__ bf16_t As[2][128][KCP];
    __shared__ bf16_t Bs[2][128][KCP];

    const int tid  = threadIdx.x;
    const int lane = tid & 31;
    const int wm   = (tid >> 5) & 3;   // 0..3 : M sub-block of 32 rows
    const int wn   = (tid >> 7);       // 0..1 : N sub-block of 64 cols

    const v8f vzero = {0.f, 0.f, 0.f, 0.f, 0.f, 0.f, 0.f, 0.f};
    v8f acc[2][4];
#pragma unroll
    for (int mt = 0; mt < 2; ++mt)
#pragma unroll
        for (int nt = 0; nt < 4; ++nt) acc[mt][nt] = vzero;

    const bf16_t* Wexp = W + (size_t)e * Dn * Dn;

    // stage one K-chunk (8 async b128 copies per thread: 4 A + 4 B)
    auto stage_chunk = [&](int c, int buf) {
        const int k0 = c * KC;
#pragma unroll
        for (int it = 0; it < 4; ++it) {
            int cc = tid + 256 * it;
            int row = cc >> 3, co = (cc & 7) * 8;
            stage16(A + (size_t)(slot0 + row) * Dn + k0 + co, &As[buf][row][co]);
            stage16(Wexp + (size_t)(nBase + row) * Dn + k0 + co, &Bs[buf][row][co]);
        }
    };

    stage_chunk(0, 0);

    for (int c = 0; c < NC; ++c) {
        const int p = c & 1;
        if (c + 1 < NC) {
            // the buffer we stage into was last read in iteration c-1,
            // whose trailing barrier guarantees all waves are done with it
            stage_chunk(c + 1, 1 - p);
            wait_async_keep8();       // chunk c complete; chunk c+1 stays in flight
        } else {
            wait_async_all();
        }
        __syncthreads();

#pragma unroll
        for (int kk = 0; kk < KC; kk += 32) {
            // A fragments per ISA 16-bit 16x32 layout:
            // lanes 0-15: k 0..7 & 16..23 ; lanes 16-31: k 8..15 & 24..31
            v16bf afrag[2];
#pragma unroll
            for (int mt = 0; mt < 2; ++mt) {
                const bf16_t* ap =
                    &As[p][wm * 32 + mt * 16 + (lane & 15)][kk + ((lane >> 4) * 8)];
                v8bf alo = *(const v8bf*)ap;
                v8bf ahi = *(const v8bf*)(ap + 16);
                afrag[mt] = __builtin_shufflevector(alo, ahi,
                    0,1,2,3,4,5,6,7,8,9,10,11,12,13,14,15);
            }
#pragma unroll
            for (int nt = 0; nt < 4; ++nt) {
                // B fragment: lane n = lane&15, same k layout as A (k 8h..8h+7 & 16+8h..16+8h+7)
                const bf16_t* bp =
                    &Bs[p][wn * 64 + nt * 16 + (lane & 15)][kk + ((lane >> 4) * 8)];
                v8bf blo = *(const v8bf*)bp;
                v8bf bhi = *(const v8bf*)(bp + 16);
                v16bf bfrag = __builtin_shufflevector(blo, bhi,
                    0,1,2,3,4,5,6,7,8,9,10,11,12,13,14,15);
#pragma unroll
                for (int mt = 0; mt < 2; ++mt)
                    acc[mt][nt] = __builtin_amdgcn_wmma_f32_16x16x32_bf16(
                        false, afrag[mt], false, bfrag, (short)0, acc[mt][nt],
                        false, false);
            }
        }
        __syncthreads();
    }

    // epilogue: C/D layout — lane col n = lane&15; VGPR j row m = (lane>>4)*8 + j
    const float* be = bias + (size_t)e * Dn;
#pragma unroll
    for (int mt = 0; mt < 2; ++mt) {
#pragma unroll
        for (int nt = 0; nt < 4; ++nt) {
            int n = nBase + wn * 64 + nt * 16 + (lane & 15);
            float bv = be[n];
#pragma unroll
            for (int j = 0; j < 8; ++j) {
                int mloc = wm * 32 + mt * 16 + (lane >> 4) * 8 + j;
                if (mloc < rows) {
                    int slot = slot0 + mloc;
                    float v = acc[mt][nt][j] + bv;
                    if (GELU) v = 0.5f * v * (1.f + erff(v * 0.70710678118654752f));
                    if (SCALE) v = v * slotW[slot];
                    Out[(size_t)slot * Dn + n] = (OutT)v;
                }
            }
        }
    }
}

// ---------- kernel 6: combine yg[slot0] + yg[slot1] (weights pre-applied) ----------
__global__ __launch_bounds__(256) void moe_combine(
    const float* __restrict__ yg, const int* __restrict__ t2s,
    float* __restrict__ out)
{
    int idx = blockIdx.x * blockDim.x + threadIdx.x;  // one float4 per thread
    int t  = idx >> 8;                                // 256 float4 per row
    int d4 = idx & 255;
    int s0 = t2s[2 * t], s1 = t2s[2 * t + 1];
    const float4* y0 = (const float4*)(yg + (size_t)s0 * Dn);
    const float4* y1 = (const float4*)(yg + (size_t)s1 * Dn);
    float4 a = y0[d4], b = y1[d4];
    float4 r; r.x = a.x + b.x; r.y = a.y + b.y; r.z = a.z + b.z; r.w = a.w + b.w;
    ((float4*)out)[idx] = r;
}

// ---------- launcher ----------
extern "C" void kernel_launch(void* const* d_in, const int* in_sizes, int n_in,
                              void* d_out, int out_size, void* d_ws, size_t ws_size,
                              hipStream_t stream) {
    const float* x  = (const float*)d_in[0];
    const float* Wr = (const float*)d_in[1];
    const float* W1 = (const float*)d_in[2];
    const float* b1 = (const float*)d_in[3];
    const float* W2 = (const float*)d_in[4];
    const float* b2 = (const float*)d_in[5];
    float* out = (float*)d_out;

    char* ws = (char*)d_ws;
    float*  topw    = (float*)(ws + OFF_TOPW);
    int*    topi    = (int*)  (ws + OFF_TOPI);
    int*    t2s     = (int*)  (ws + OFF_T2S);
    float*  slotW   = (float*)(ws + OFF_SW);
    int*    counts  = (int*)  (ws + OFF_COUNTS);
    int*    cursors = (int*)  (ws + OFF_CURSORS);
    int*    bases   = (int*)  (ws + OFF_BASES);
    int*    blkExp  = (int*)  (ws + OFF_BEXP);
    int*    blkSlot = (int*)  (ws + OFF_BSLOT);
    int*    blkRows = (int*)  (ws + OFF_BROWS);
    bf16_t* xg      = (bf16_t*)(ws + OFF_XG);
    bf16_t* hg      = (bf16_t*)(ws + OFF_HG);
    float*  yg      = (float*)(ws + OFF_YG);
    bf16_t* W1b     = (bf16_t*)(ws + OFF_W1B);
    bf16_t* W2b     = (bf16_t*)(ws + OFF_W2B);

    moe_init<<<1, 64, 0, stream>>>(counts, cursors);

    // one-time weight downcast (E*D*D = 8.4M elems each, 8 elems/thread)
    constexpr int CVTB = (En * Dn * Dn) / (256 * 8);
    moe_cvtw<<<CVTB, 256, 0, stream>>>(W1, W1b);
    moe_cvtw<<<CVTB, 256, 0, stream>>>(W2, W2b);

    moe_router<<<Tn / 8, 256, 0, stream>>>(x, Wr, topw, topi, counts);
    moe_scan<<<1, 32, 0, stream>>>(counts, bases, blkExp, blkSlot, blkRows);
    moe_gather<<<NSLOTS / 8, 256, 0, stream>>>(x, topi, topw, bases, cursors, t2s, slotW, xg);

    dim3 ggrid(MAXB, Dn / 128);
    moe_gemm<true,  false, bf16_t><<<ggrid, 256, 0, stream>>>(
        xg, W1b, b1, blkExp, blkSlot, blkRows, slotW, hg);
    moe_gemm<false, true,  float ><<<ggrid, 256, 0, stream>>>(
        hg, W2b, b2, blkExp, blkSlot, blkRows, slotW, yg);

    moe_combine<<<(Tn * (Dn / 4)) / 256, 256, 0, stream>>>(yg, t2s, out);
}
